// CausalMultiHeadSelfAttention_75350906241314
// MI455X (gfx1250) — compile-verified, matrix operand order checked by reading, not run on hardware
//
#include <hip/hip_runtime.h>

typedef unsigned short u16;
typedef __attribute__((ext_vector_type(16))) __bf16 v16bf;
typedef __attribute__((ext_vector_type(8)))  float  v8f;
typedef __attribute__((ext_vector_type(4))) unsigned int u32x4;
typedef __attribute__((ext_vector_type(8))) int i32x8;
typedef __attribute__((ext_vector_type(4))) int i32x4;

constexpr int Bsz = 2;
constexpr int S   = 2048;
constexpr int Dm  = 1024;
constexpr int NH  = 16;
constexpr int Dh  = 64;

#ifndef __has_builtin
#define __has_builtin(x) 0
#endif
#if __has_builtin(__builtin_amdgcn_tensor_load_to_lds) && \
    __has_builtin(__builtin_amdgcn_s_wait_tensorcnt) && defined(__AMDGCN__)
#define HAVE_TDM 1
#else
#define HAVE_TDM 0
#if defined(__AMDGCN__)
#warning "gfx1250 TDM builtins unavailable: attention kernel uses global-load fallback"
#endif
#endif

union FragCvt { v16bf bf; uint4 u4[2]; };

__device__ inline u16 f2bf(float f) {
  unsigned u = __float_as_uint(f);
  return (u16)((u + 0x7FFFu + ((u >> 16) & 1u)) >> 16);
}

__device__ inline v8f zero8() {
  v8f z = {0.f, 0.f, 0.f, 0.f, 0.f, 0.f, 0.f, 0.f};
  return z;
}

// 16x32 bf16 A-fragment (and B-fragment when the source is "B^T row-major").
// ISA 7.12.2: lane<16 holds row=lane, K=k0+{0..7,16..23}; lane>=16 row=lane-16,
// K=k0+{8..15,24..31}.
__device__ inline v16bf ld_frag(const u16* base, int ld, int row0, int k0, int lane) {
  const u16* p = base + (size_t)(row0 + (lane & 15)) * ld + k0 + ((lane >> 4) << 3);
  FragCvt f;
  f.u4[0] = *reinterpret_cast<const uint4*>(p);
  f.u4[1] = *reinterpret_cast<const uint4*>(p + 16);
  return f.bf;
}

#define WMMA_BF16(a, b, c) \
  __builtin_amdgcn_wmma_f32_16x16x32_bf16(false, (a), false, (b), (short)0, (c), false, false)

#if HAVE_TDM
// Build a TDM D# (ISA ch.8) for a 2D (or 1D with tile_d1==0) bf16 tile load
// and issue TENSOR_LOAD_TO_LDS. All inputs must be wave-uniform.
__device__ inline void tdm_load_tile(unsigned lds_off, const void* gaddr,
                                     unsigned tensor_d0, unsigned tensor_d1,
                                     unsigned tile_d0, unsigned tile_d1,
                                     unsigned long long stride0) {
  unsigned long long ga = (unsigned long long)(uintptr_t)gaddr;
  // Group 0: count=1 | lds_addr | global_addr[56:0] | type=2 ("image")
  unsigned g0w0 = 1u;
  unsigned g0w1 = lds_off;
  unsigned g0w2 = (unsigned)ga;
  unsigned g0w3 = (unsigned)((ga >> 32) & 0x1FFFFFFu) | (2u << 30);
  // Group 1: data_size=1 (2B elems); tensor/tile dims; dim0 stride.
  unsigned g1w0 = (1u << 16);
  unsigned g1w1 = (tensor_d0 & 0xFFFFu) << 16;
  unsigned g1w2 = ((tensor_d0 >> 16) & 0xFFFFu) | ((tensor_d1 & 0xFFFFu) << 16);
  unsigned g1w3 = ((tensor_d1 >> 16) & 0xFFFFu) | ((tile_d0 & 0xFFFFu) << 16);
  unsigned g1w4 = (tile_d1 & 0xFFFFu);
  unsigned g1w5 = (unsigned)(stride0 & 0xFFFFFFFFull);
  unsigned g1w6 = (unsigned)((stride0 >> 32) & 0xFFFFull);
  unsigned g1w7 = 0u;
#define RFL(x) __builtin_amdgcn_readfirstlane((int)(x))
  u32x4 g0 = {(unsigned)RFL(g0w0), (unsigned)RFL(g0w1),
              (unsigned)RFL(g0w2), (unsigned)RFL(g0w3)};
  i32x8 g1 = {RFL(g1w0), RFL(g1w1), RFL(g1w2), RFL(g1w3),
              RFL(g1w4), RFL(g1w5), RFL(g1w6), RFL(g1w7)};
#undef RFL
  i32x4 gz = {0, 0, 0, 0};
#if __clang_major__ >= 23
  i32x8 gz8 = {0, 0, 0, 0, 0, 0, 0, 0};
  __builtin_amdgcn_tensor_load_to_lds(g0, g1, gz, gz, gz8, 0);
#else
  __builtin_amdgcn_tensor_load_to_lds(g0, g1, gz, gz, 0);
#endif
}
#endif  // HAVE_TDM

// ---------------------------------------------------------------------------
// Kernel 1: f32 -> bf16 cast, vectorized (float4 in, 4x bf16 packed out).
// All array sizes here are multiples of 4.
// ---------------------------------------------------------------------------
__global__ void cast_f32_bf16_kernel(const float4* __restrict__ src, uint2* __restrict__ dst,
                                     int n4) {
  int i = blockIdx.x * blockDim.x + threadIdx.x;
  int stride = gridDim.x * blockDim.x;
  for (; i < n4; i += stride) {
    float4 v = src[i];
    uint2 o;
    o.x = (unsigned)f2bf(v.x) | ((unsigned)f2bf(v.y) << 16);
    o.y = (unsigned)f2bf(v.z) | ((unsigned)f2bf(v.w) << 16);
    dst[i] = o;
  }
}

// ---------------------------------------------------------------------------
// Kernel 2: fused QKV projection + RoPE, software-pipelined, 32x16x{Q,K,V}
// per wave (6 WMMA / 5 fragment loads per k-step).
// grid = (B*S/32, NH), block = 128 (4 waves; wave w -> head cols [w*16,w*16+16)).
// ---------------------------------------------------------------------------
__global__ __launch_bounds__(128) void qkv_rope_kernel(
    const u16* __restrict__ Xbf,
    const u16* __restrict__ Wq, const u16* __restrict__ Wk, const u16* __restrict__ Wv,
    u16* __restrict__ Qh, u16* __restrict__ Kh, u16* __restrict__ Vt) {
  const int wave = threadIdx.x >> 5;
  const int lane = threadIdx.x & 31;
  const int n    = lane & 15;
  const int half = lane >> 4;

  const int r0 = blockIdx.x * 32;
  const int h  = blockIdx.y;
  const int b  = r0 >> 11;
  const int s0 = r0 & (S - 1);
  const int e0 = h * Dh + wave * 16;

  v8f cq[2], ck[2], cv[2];
#pragma unroll
  for (int ri = 0; ri < 2; ++ri) { cq[ri] = zero8(); ck[ri] = zero8(); cv[ri] = zero8(); }

  // Software pipeline: preload k-step fragments one iteration ahead.
  v16bf a0c = ld_frag(Xbf, Dm, r0,      0, lane);
  v16bf a1c = ld_frag(Xbf, Dm, r0 + 16, 0, lane);
  v16bf bqc = ld_frag(Wq,  Dm, e0, 0, lane);
  v16bf bkc = ld_frag(Wk,  Dm, e0, 0, lane);
  v16bf bvc = ld_frag(Wv,  Dm, e0, 0, lane);
  for (int k0 = 0; k0 < Dm - 32; k0 += 32) {
    int kn = k0 + 32;
    v16bf a0n = ld_frag(Xbf, Dm, r0,      kn, lane);
    v16bf a1n = ld_frag(Xbf, Dm, r0 + 16, kn, lane);
    v16bf bqn = ld_frag(Wq,  Dm, e0, kn, lane);
    v16bf bkn = ld_frag(Wk,  Dm, e0, kn, lane);
    v16bf bvn = ld_frag(Wv,  Dm, e0, kn, lane);
    cq[0] = WMMA_BF16(a0c, bqc, cq[0]); cq[1] = WMMA_BF16(a1c, bqc, cq[1]);
    ck[0] = WMMA_BF16(a0c, bkc, ck[0]); ck[1] = WMMA_BF16(a1c, bkc, ck[1]);
    cv[0] = WMMA_BF16(a0c, bvc, cv[0]); cv[1] = WMMA_BF16(a1c, bvc, cv[1]);
    a0c = a0n; a1c = a1n; bqc = bqn; bkc = bkn; bvc = bvn;
  }
  cq[0] = WMMA_BF16(a0c, bqc, cq[0]); cq[1] = WMMA_BF16(a1c, bqc, cq[1]);
  ck[0] = WMMA_BF16(a0c, bkc, ck[0]); ck[1] = WMMA_BF16(a1c, bkc, ck[1]);
  cv[0] = WMMA_BF16(a0c, bvc, cv[0]); cv[1] = WMMA_BF16(a1c, bvc, cv[1]);

  // RoPE on Q and K. C layout: lane holds col n, rows r+8*half.
  const int dl = wave * 16 + n;
  const float freq = __powf(10000.0f, -(float)(dl & ~1) * (1.0f / 64.0f));
  const bool even = (dl & 1) == 0;
  const size_t headQK = ((size_t)(b * NH + h)) * S * Dh;
  const size_t headV  = ((size_t)(b * NH + h)) * Dh * S;
#pragma unroll
  for (int ri = 0; ri < 2; ++ri) {
#pragma unroll
    for (int e = 0; e < 8; ++e) {
      float pos = (float)(s0 + ri * 16 + e + 8 * half);
      float sn, cs;
      __sincosf(pos * freq, &sn, &cs);
      float q  = cq[ri][e];
      float qo = __shfl_xor(q, 1, 32);
      cq[ri][e] = even ? (q * cs - qo * sn) : (qo * sn + q * cs);
      float k  = ck[ri][e];
      float ko = __shfl_xor(k, 1, 32);
      ck[ri][e] = even ? (k * cs - ko * sn) : (ko * sn + k * cs);
    }
#pragma unroll
    for (int e = 0; e < 8; ++e) {
      int srow = s0 + ri * 16 + e + 8 * half;
      Qh[headQK + (size_t)srow * Dh + dl] = f2bf(cq[ri][e]);
      Kh[headQK + (size_t)srow * Dh + dl] = f2bf(ck[ri][e]);
      Vt[headV + (size_t)dl * S + srow]   = f2bf(cv[ri][e]);
    }
  }
}

// ---------------------------------------------------------------------------
// Kernel 3: causal flash attention. grid = (B*NH, S/64), block = 128 (4 waves,
// 16 q-rows each). K/V tiles (32 keys) are staged into LDS by the Tensor Data
// Mover, double-buffered, shared by all 4 waves; uniform trip count (causal
// masking makes extra key blocks no-ops).
// ---------------------------------------------------------------------------
__global__ __launch_bounds__(128) void attn_kernel(
    const u16* __restrict__ Qh, const u16* __restrict__ Kh, const u16* __restrict__ Vt,
    u16* __restrict__ Obf) {
  __shared__ u16 kbuf[2][32 * Dh];   // [buf][key_local][d]
  __shared__ u16 vbuf[2][Dh * 32];   // [buf][d][key_local]
  __shared__ u16 pstage[4][16 * 32];

  const int wave = threadIdx.x >> 5;
  const int lane = threadIdx.x & 31;
  const int n    = lane & 15;
  const int half = lane >> 4;

  const int bh  = blockIdx.x;  // b*NH + h
  const int b   = bh >> 4;
  const int h   = bh & 15;
  const int bq0 = blockIdx.y * 64;
  const int q0  = bq0 + wave * 16;

  const u16* Qp = Qh + (size_t)bh * S * Dh;
  const u16* Kp = Kh + (size_t)bh * S * Dh;
  const u16* Vp = Vt + (size_t)bh * Dh * S;

  const v16bf a0 = ld_frag(Qp, Dh, q0, 0, lane);
  const v16bf a1 = ld_frag(Qp, Dh, q0, 32, lane);

  v8f o0 = zero8(), o1 = zero8(), o2 = zero8(), o3 = zero8();
  float mrow[8], lrow[8];
#pragma unroll
  for (int e = 0; e < 8; ++e) { mrow[e] = -1e30f; lrow[e] = 0.f; }

  const float scale = 0.125f;       // 1/sqrt(64)
  const int nIter = bq0 / 32 + 2;   // block-uniform: keys < bq0+64

#if HAVE_TDM
  if (wave == 0) {
    // K tile: 32 rows x 64 d, fully contiguous -> 1D copy of 2048 elems.
    tdm_load_tile((unsigned)(uintptr_t)&kbuf[0][0], Kp,
                  (unsigned)(S * Dh), 0u, 2048u, 0u, (unsigned long long)(S * Dh));
    // V tile: 64 rows (d) x 32 keys from Vt[Dh][S].
    tdm_load_tile((unsigned)(uintptr_t)&vbuf[0][0], Vp,
                  (unsigned)S, (unsigned)Dh, 32u, (unsigned)Dh, (unsigned long long)S);
  }
#endif

  for (int i = 0; i < nIter; ++i) {
    const int kb = i * 32;
    const int pb = i & 1;

#if HAVE_TDM
    __syncthreads();  // all reads of buffer pb^1 (iteration i-1) are done
    if (wave == 0) {
      if (i + 1 < nIter) {
        tdm_load_tile((unsigned)(uintptr_t)&kbuf[pb ^ 1][0], Kp + (size_t)(i + 1) * 32 * Dh,
                      (unsigned)(S * Dh), 0u, 2048u, 0u, (unsigned long long)(S * Dh));
        tdm_load_tile((unsigned)(uintptr_t)&vbuf[pb ^ 1][0], Vp + (size_t)(i + 1) * 32,
                      (unsigned)S, (unsigned)Dh, 32u, (unsigned)Dh, (unsigned long long)S);
        __builtin_amdgcn_s_wait_tensorcnt(2);  // iteration i's pair complete
      } else {
        __builtin_amdgcn_s_wait_tensorcnt(0);
      }
    }
    __syncthreads();  // publish buffer pb
    const u16* Ksrc = &kbuf[pb][0]; const int kld = Dh; const int krow = 0;
    const u16* Vsrc = &vbuf[pb][0]; const int vld = 32; const int vk0 = 0;
#else
    const u16* Ksrc = Kp + (size_t)kb * Dh; const int kld = Dh; const int krow = 0;
    const u16* Vsrc = Vp;                   const int vld = S;  const int vk0 = kb;
    if (kb + 48 <= S)
      __builtin_prefetch((const void*)(Kp + (size_t)(kb + 32) * Dh), 0, 1);
#endif

    // Scores: 16 q-rows x 32 keys, K-dim 64.
    v8f s1 = zero8(), s2 = zero8();
    s1 = WMMA_BF16(a0, ld_frag(Ksrc, kld, krow,      0,  lane), s1);
    s1 = WMMA_BF16(a1, ld_frag(Ksrc, kld, krow,      32, lane), s1);
    s2 = WMMA_BF16(a0, ld_frag(Ksrc, kld, krow + 16, 0,  lane), s2);
    s2 = WMMA_BF16(a1, ld_frag(Ksrc, kld, krow + 16, 32, lane), s2);

    float alpha[8];
#pragma unroll
    for (int e = 0; e < 8; ++e) {
      int qrow = q0 + e + 8 * half;
      float v1 = (kb + n      <= qrow) ? s1[e] * scale : -1e30f;
      float v2 = (kb + 16 + n <= qrow) ? s2[e] * scale : -1e30f;

      float mx = fmaxf(v1, v2);
      mx = fmaxf(mx, __shfl_xor(mx, 1, 32));
      mx = fmaxf(mx, __shfl_xor(mx, 2, 32));
      mx = fmaxf(mx, __shfl_xor(mx, 4, 32));
      mx = fmaxf(mx, __shfl_xor(mx, 8, 32));

      float mnew = fmaxf(mrow[e], mx);
      alpha[e] = __expf(mrow[e] - mnew);
      mrow[e] = mnew;

      float p1 = __expf(v1 - mnew);
      float p2 = __expf(v2 - mnew);
      float sum = p1 + p2;
      sum += __shfl_xor(sum, 1, 32);
      sum += __shfl_xor(sum, 2, 32);
      sum += __shfl_xor(sum, 4, 32);
      sum += __shfl_xor(sum, 8, 32);
      lrow[e] = lrow[e] * alpha[e] + sum;

      pstage[wave][(e + 8 * half) * 32 + n]      = f2bf(p1);
      pstage[wave][(e + 8 * half) * 32 + n + 16] = f2bf(p2);
    }

#pragma unroll
    for (int e = 0; e < 8; ++e) {
      o0[e] *= alpha[e]; o1[e] *= alpha[e]; o2[e] *= alpha[e]; o3[e] *= alpha[e];
    }

    asm volatile("s_wait_dscnt 0" ::: "memory");

    FragCvt pf;
    const u16* pp = &pstage[wave][n * 32 + half * 8];
    pf.u4[0] = *reinterpret_cast<const uint4*>(pp);
    pf.u4[1] = *reinterpret_cast<const uint4*>(pp + 16);

    o0 = WMMA_BF16(pf.bf, ld_frag(Vsrc, vld, 0,  vk0, lane), o0);
    o1 = WMMA_BF16(pf.bf, ld_frag(Vsrc, vld, 16, vk0, lane), o1);
    o2 = WMMA_BF16(pf.bf, ld_frag(Vsrc, vld, 32, vk0, lane), o2);
    o3 = WMMA_BF16(pf.bf, ld_frag(Vsrc, vld, 48, vk0, lane), o3);
  }

#pragma unroll
  for (int e = 0; e < 8; ++e) {
    float inv = 1.0f / lrow[e];
    o0[e] *= inv; o1[e] *= inv; o2[e] *= inv; o3[e] *= inv;
  }

#pragma unroll
  for (int e = 0; e < 8; ++e) {
    int srow = q0 + e + 8 * half;
    size_t base = ((size_t)(b * S + srow)) * Dm + h * Dh;
    Obf[base + 0  + n] = f2bf(o0[e]);
    Obf[base + 16 + n] = f2bf(o1[e]);
    Obf[base + 32 + n] = f2bf(o2[e]);
    Obf[base + 48 + n] = f2bf(o3[e]);
  }
}

// ---------------------------------------------------------------------------
// Kernel 4: output projection, 32x32 C tile per wave (4 WMMA / 4 loads),
// software-pipelined. grid = (B*S/32, Dm/128), block = 128.
// ---------------------------------------------------------------------------
__global__ __launch_bounds__(128) void out_proj_kernel(
    const u16* __restrict__ Obf, const u16* __restrict__ Wo, float* __restrict__ out) {
  const int wave = threadIdx.x >> 5;
  const int lane = threadIdx.x & 31;
  const int n    = lane & 15;
  const int half = lane >> 4;

  const int r0 = blockIdx.x * 32;
  const int e0 = blockIdx.y * 128 + wave * 32;

  v8f c00 = zero8(), c01 = zero8(), c10 = zero8(), c11 = zero8();

  v16bf a0c = ld_frag(Obf, Dm, r0,      0, lane);
  v16bf a1c = ld_frag(Obf, Dm, r0 + 16, 0, lane);
  v16bf b0c = ld_frag(Wo,  Dm, e0,      0, lane);
  v16bf b1c = ld_frag(Wo,  Dm, e0 + 16, 0, lane);
  for (int k0 = 0; k0 < Dm - 32; k0 += 32) {
    int kn = k0 + 32;
    v16bf a0n = ld_frag(Obf, Dm, r0,      kn, lane);
    v16bf a1n = ld_frag(Obf, Dm, r0 + 16, kn, lane);
    v16bf b0n = ld_frag(Wo,  Dm, e0,      kn, lane);
    v16bf b1n = ld_frag(Wo,  Dm, e0 + 16, kn, lane);
    c00 = WMMA_BF16(a0c, b0c, c00); c01 = WMMA_BF16(a0c, b1c, c01);
    c10 = WMMA_BF16(a1c, b0c, c10); c11 = WMMA_BF16(a1c, b1c, c11);
    a0c = a0n; a1c = a1n; b0c = b0n; b1c = b1n;
  }
  c00 = WMMA_BF16(a0c, b0c, c00); c01 = WMMA_BF16(a0c, b1c, c01);
  c10 = WMMA_BF16(a1c, b0c, c10); c11 = WMMA_BF16(a1c, b1c, c11);

#pragma unroll
  for (int e = 0; e < 8; ++e) {
    size_t row0 = (size_t)(r0 + e + 8 * half) * Dm;
    size_t row1 = (size_t)(r0 + 16 + e + 8 * half) * Dm;
    out[row0 + e0 + n]      = c00[e];
    out[row0 + e0 + 16 + n] = c01[e];
    out[row1 + e0 + n]      = c10[e];
    out[row1 + e0 + 16 + n] = c11[e];
  }
}

// ---------------------------------------------------------------------------
// Launch
// ---------------------------------------------------------------------------
extern "C" void kernel_launch(void* const* d_in, const int* in_sizes, int n_in,
                              void* d_out, int out_size, void* d_ws, size_t ws_size,
                              hipStream_t stream) {
  const float* x  = (const float*)d_in[0];
  const float* Wq = (const float*)d_in[1];
  const float* Wk = (const float*)d_in[2];
  const float* Wv = (const float*)d_in[3];
  const float* Wo = (const float*)d_in[4];
  float* out = (float*)d_out;

  char* ws = (char*)d_ws;
  constexpr size_t XB = (size_t)Bsz * S * Dm * sizeof(u16);      // 8 MiB
  constexpr size_t WB = (size_t)Dm * Dm * sizeof(u16);           // 2 MiB
  constexpr size_t HB = (size_t)Bsz * NH * S * Dh * sizeof(u16); // 8 MiB

  u16* Xbf = (u16*)(ws);
  u16* Wqb = (u16*)(ws + XB);
  u16* Wkb = (u16*)(ws + XB + WB);
  u16* Wvb = (u16*)(ws + XB + 2 * WB);
  u16* Wob = (u16*)(ws + XB + 3 * WB);
  u16* Qh  = (u16*)(ws + XB + 4 * WB);
  u16* Kh  = (u16*)(ws + XB + 4 * WB + HB);
  u16* Vt  = (u16*)(ws + XB + 4 * WB + 2 * HB);
  u16* Obf = (u16*)(ws + XB + 4 * WB + 3 * HB);

  const int nX = Bsz * S * Dm;
  const int nW = Dm * Dm;
  cast_f32_bf16_kernel<<<1024, 256, 0, stream>>>((const float4*)x,  (uint2*)Xbf, nX / 4);
  cast_f32_bf16_kernel<<<512,  256, 0, stream>>>((const float4*)Wq, (uint2*)Wqb, nW / 4);
  cast_f32_bf16_kernel<<<512,  256, 0, stream>>>((const float4*)Wk, (uint2*)Wkb, nW / 4);
  cast_f32_bf16_kernel<<<512,  256, 0, stream>>>((const float4*)Wv, (uint2*)Wvb, nW / 4);
  cast_f32_bf16_kernel<<<512,  256, 0, stream>>>((const float4*)Wo, (uint2*)Wob, nW / 4);

  qkv_rope_kernel<<<dim3(Bsz * S / 32, NH), 128, 0, stream>>>(Xbf, Wqb, Wkb, Wvb, Qh, Kh, Vt);
  attn_kernel<<<dim3(Bsz * NH, S / 64), 128, 0, stream>>>(Qh, Kh, Vt, Obf);
  out_proj_kernel<<<dim3(Bsz * S / 32, Dm / 128), 128, 0, stream>>>(Obf, Wob, out);
}
